// NSF_AR_Block_26603027431457
// MI455X (gfx1250) — matrix operand order repaired and checked by reading, not compiled, not run on hardware
//
#include <hip/hip_runtime.h>
#include <hip/hip_bf16.h>

// ---------------------------------------------------------------------------
// Types for WMMA fragments (gfx1250, wave32)
// ---------------------------------------------------------------------------
typedef __attribute__((ext_vector_type(16))) _Float16 v16h;
typedef __attribute__((ext_vector_type(8)))  _Float16 v8h;
typedef __attribute__((ext_vector_type(8)))  float    v8f;

union H16 { v16h v; v8h h[2]; };
union F8  { v8f  v; float f[8]; };

// Problem constants
#define BATCH   65536
#define IN_DIM  8
#define COND_DIM 6
#define NN      256          // hidden width
#define KPARAM  5
#define NPARAM  14           // 3*K - 1
#define MTILE   64

// Packed f16 weight layout per dim (offsets in halves)
#define OFF_B0   0           // [256][32]  (w0 cols | cw0 cols | 0-pad)
#define OFF_W1   8192        // [256][256]
#define OFF_CW1  73728       // [256][32]
#define OFF_W2   81920       // [256][256]
#define OFF_CW2  147456      // [256][32]
#define OFF_W3   155648      // [16][256]  (rows 14..15 zero)
#define OFF_CW3  159744      // [16][32]
#define WPD      160256      // halves per dim
#define BPD      784         // f32 biases per dim: 256+256+256+16

// ---------------------------------------------------------------------------
// Branch-free native-math helpers (v_exp_f32 / v_log_f32 / v_rcp_f32)
// ---------------------------------------------------------------------------
__device__ __forceinline__ float fast_tanhf(float v) {
  // tanh(v) = 1 - 2 / (exp(2v) + 1); saturates correctly at +-inf, no branches
  const float e = __expf(2.0f * v);
  return 1.0f - 2.0f * __builtin_amdgcn_rcpf(e + 1.0f);
}

__device__ __forceinline__ float fast_softplusf(float v) {
  // softplus(v) = max(v,0) + log(1 + exp(-|v|)) ; stable, branch-free
  return fmaxf(v, 0.0f) + __logf(1.0f + __expf(-fabsf(v)));
}

// ---------------------------------------------------------------------------
// Fragment loaders (layouts per CDNA5 ISA 7.12.2)
// ---------------------------------------------------------------------------
// A (16x32 f16): lanes 0-15 -> M=lane, K chunks [kb, kb+8) and [kb+16, kb+24)
//                lanes16-31 -> M=lane-16, K chunks [kb+8, kb+16) and [kb+24, kb+32)
__device__ __forceinline__ v16h frag_a_lds(const _Float16* base, int stride,
                                           int mb, int kb, int lane) {
  const int m  = mb + (lane & 15);
  const int hi = lane >> 4;
  const _Float16* p = base + m * stride + kb + hi * 8;
  H16 u;
  u.h[0] = *(const v8h*)p;
  u.h[1] = *(const v8h*)(p + 16);
  return u.v;
}

// B (32x16 f16): lanes 0-15 -> N=lane, K chunks [kb, kb+8) and [kb+16, kb+24)
//                lanes16-31 -> N=lane-16, K chunks [kb+8, kb+16) and [kb+24, kb+32)
// Packed weights are [n][k] row-major, so this is two 16-byte runs.
__device__ __forceinline__ v16h frag_b_glb(const _Float16* __restrict__ W, int kstride,
                                           int nb, int kb, int lane) {
  const int n  = nb + (lane & 15);
  const int hi = lane >> 4;
  const _Float16* p = W + (size_t)(n) * kstride + kb + hi * 8;
  H16 u;
  u.h[0] = *(const v8h*)p;
  u.h[1] = *(const v8h*)(p + 16);
  return u.v;
}

__device__ __forceinline__ v8f wmma_f16(v16h a, v16h b, v8f c) {
  return __builtin_amdgcn_wmma_f32_16x16x32_f16(false, a, false, b, (short)0, c, false, false);
}

// ---------------------------------------------------------------------------
// One 256->256 hidden layer: C = tanh(A*W^T + b + cond*CW^T), A/out in LDS
// ---------------------------------------------------------------------------
__device__ __forceinline__ void layer_256x256(
    const _Float16* __restrict__ W,   // [256][256] f16
    const _Float16* __restrict__ CW,  // [256][32]  f16 (cond cols 0..5, rest 0)
    const float*    __restrict__ bias,
    _Float16 (*As)[264],              // activations [64][264], K=0..255 valid
    _Float16 (*Cs)[32],               // cond tile   [64][32]
    int lane, int wm, int wn)
{
  const int nbase = wn * 128;
  const int nloc  = lane & 15;
  const int hi    = lane >> 4;

  F8 acc[8];
  #pragma unroll
  for (int t = 0; t < 8; ++t) {
    const float bv = bias[nbase + t * 16 + nloc];
    #pragma unroll
    for (int r = 0; r < 8; ++r) acc[t].f[r] = bv;
  }

  // main K loop over the 256-wide activation
  for (int kk = 0; kk < 8; ++kk) {
    const v16h a = frag_a_lds(&As[0][0], 264, wm * 16, kk * 32, lane);
    #pragma unroll
    for (int t = 0; t < 8; ++t) {
      const v16h b = frag_b_glb(W, 256, nbase + t * 16, kk * 32, lane);
      acc[t].v = wmma_f16(a, b, acc[t].v);
    }
  }
  // conditioning K-tile (cond padded to K=32)
  {
    const v16h a = frag_a_lds(&Cs[0][0], 32, wm * 16, 0, lane);
    #pragma unroll
    for (int t = 0; t < 8; ++t) {
      const v16h b = frag_b_glb(CW, 32, nbase + t * 16, 0, lane);
      acc[t].v = wmma_f16(a, b, acc[t].v);
    }
  }

  __syncthreads();                 // all waves finished reading As
  #pragma unroll
  for (int t = 0; t < 8; ++t) {
    #pragma unroll
    for (int r = 0; r < 8; ++r) {
      const int m = wm * 16 + hi * 8 + r;       // C layout: vgpr r -> M=r (+8 for hi lanes)
      As[m][nbase + t * 16 + nloc] = (_Float16)fast_tanhf(acc[t].f[r]);
    }
  }
  __syncthreads();
}

// ---------------------------------------------------------------------------
// Fused per-dim MLP: grid.x = BATCH/64, grid.y = 7 (dim = y+1), 256 threads
// ---------------------------------------------------------------------------
__global__ __launch_bounds__(256)
void nsf_mlp_kernel(const float* __restrict__ x, const float* __restrict__ cond,
                    const _Float16* __restrict__ wpack, const float* __restrict__ bpack,
                    float* __restrict__ p_buf)
{
  const int dd  = blockIdx.y;       // 0..6
  const int dim = dd + 1;
  const int m0  = blockIdx.x * MTILE;

  const _Float16* wb = wpack + (size_t)dd * WPD;
  const float*    bb = bpack + (size_t)dd * BPD;

  __shared__ __align__(16) _Float16 As[MTILE][264];
  __shared__ __align__(16) _Float16 A0s[MTILE][32];
  __shared__ __align__(16) _Float16 Cs[MTILE][32];

  const int tid  = threadIdx.x;
  const int lane = tid & 31;
  const int wave = tid >> 5;
  const int wm   = wave & 3;        // M subtile (16 rows)
  const int wn   = wave >> 2;       // N half   (128 cols)
  const int nloc = lane & 15;
  const int hi   = lane >> 4;

  // Build layer-0 input tile ([x(:,:dim) | cond | 0]) and cond tile ([cond | 0])
  for (int t = tid; t < MTILE * 32; t += 256) {
    const int r = t >> 5, c = t & 31;
    const int m = m0 + r;
    float a0v = 0.f, cv = 0.f;
    if (c < dim)            a0v = x[(size_t)m * IN_DIM + c];
    else if (c < dim + 6)   a0v = cond[(size_t)m * COND_DIM + (c - dim)];
    if (c < 6)              cv  = cond[(size_t)m * COND_DIM + c];
    A0s[r][c] = (_Float16)a0v;
    Cs[r][c]  = (_Float16)cv;
  }
  __syncthreads();

  // ---- layer 0: K=32 (x|cond|0), N=256, tanh -> As
  {
    const _Float16* B0 = wb + OFF_B0;
    const float*    b0 = bb;
    const v16h a = frag_a_lds(&A0s[0][0], 32, wm * 16, 0, lane);
    #pragma unroll
    for (int t = 0; t < 8; ++t) {
      const int nb = wn * 128 + t * 16;
      F8 c;
      const float bv = b0[nb + nloc];
      #pragma unroll
      for (int r = 0; r < 8; ++r) c.f[r] = bv;
      const v16h b = frag_b_glb(B0, 32, nb, 0, lane);
      c.v = wmma_f16(a, b, c.v);
      #pragma unroll
      for (int r = 0; r < 8; ++r) {
        const int m = wm * 16 + hi * 8 + r;
        As[m][nb + nloc] = (_Float16)fast_tanhf(c.f[r]);
      }
    }
  }
  __syncthreads();

  // ---- layers 1 and 2: 256 -> 256 with cond + tanh
  layer_256x256(wb + OFF_W1, wb + OFF_CW1, bb + 256, As, Cs, lane, wm, wn);
  layer_256x256(wb + OFF_W2, wb + OFF_CW2, bb + 512, As, Cs, lane, wm, wn);

  // ---- layer 3: 256 -> 14 (padded to 16), no tanh; waves 0..3 only (wave-uniform)
  if (wn == 0) {
    const _Float16* W3  = wb + OFF_W3;
    const _Float16* CW3 = wb + OFF_CW3;
    const float*    b3  = bb + 768;   // 16 entries, rows 14..15 are zero
    F8 acc;
    const float bv = b3[nloc];
    #pragma unroll
    for (int r = 0; r < 8; ++r) acc.f[r] = bv;
    for (int kk = 0; kk < 8; ++kk) {
      const v16h a = frag_a_lds(&As[0][0], 264, wm * 16, kk * 32, lane);
      const v16h b = frag_b_glb(W3, 256, 0, kk * 32, lane);
      acc.v = wmma_f16(a, b, acc.v);
    }
    {
      const v16h a = frag_a_lds(&Cs[0][0], 32, wm * 16, 0, lane);
      const v16h b = frag_b_glb(CW3, 32, 0, 0, lane);
      acc.v = wmma_f16(a, b, acc.v);
    }
    #pragma unroll
    for (int r = 0; r < 8; ++r) {
      const int m = m0 + wm * 16 + hi * 8 + r;
      p_buf[((size_t)dd * BATCH + m) * 16 + nloc] = acc.f[r];
    }
  }
}

// ---------------------------------------------------------------------------
// Rational-quadratic spline (exact reference math), one thread per (row, dim)
// ---------------------------------------------------------------------------
__global__ __launch_bounds__(256)
void nsf_spline_kernel(const float* __restrict__ x, const float* __restrict__ init_param,
                       const float* __restrict__ p_buf, float* __restrict__ out)
{
  const int gid = blockIdx.x * 256 + threadIdx.x;
  if (gid >= BATCH * IN_DIM) return;
  const int d = gid & 7;
  const int m = gid >> 3;

  float p[NPARAM];
  if (d == 0) {
    #pragma unroll
    for (int i = 0; i < NPARAM; ++i) p[i] = init_param[i];
  } else {
    const float* pp = p_buf + (((size_t)(d - 1)) * BATCH + m) * 16;
    #pragma unroll
    for (int i = 0; i < NPARAM; ++i) p[i] = pp[i];
  }
  const float xv = x[(size_t)m * IN_DIM + d];

  // _forward: w = 2*softmax(uw), h = 2*softmax(uh), dd = softplus(ud)
  float w2[5], h2[5], dd4[4];
  {
    float mx = p[0];
    #pragma unroll
    for (int i = 1; i < 5; ++i) mx = fmaxf(mx, p[i]);
    float s = 0.f;
    #pragma unroll
    for (int i = 0; i < 5; ++i) { w2[i] = __expf(p[i] - mx); s += w2[i]; }
    const float inv = 2.0f / s;
    #pragma unroll
    for (int i = 0; i < 5; ++i) w2[i] *= inv;
  }
  {
    float mx = p[5];
    #pragma unroll
    for (int i = 1; i < 5; ++i) mx = fmaxf(mx, p[5 + i]);
    float s = 0.f;
    #pragma unroll
    for (int i = 0; i < 5; ++i) { h2[i] = __expf(p[5 + i] - mx); s += h2[i]; }
    const float inv = 2.0f / s;
    #pragma unroll
    for (int i = 0; i < 5; ++i) h2[i] *= inv;
  }
  #pragma unroll
  for (int i = 0; i < 4; ++i) dd4[i] = fast_softplusf(p[10 + i]);

  // _rqs
  const float CONSTD = __logf(__expf(1.0f - 0.001f) - 1.0f);
  float dpad[6];
  dpad[0] = CONSTD; dpad[5] = CONSTD;
  #pragma unroll
  for (int i = 0; i < 4; ++i) dpad[1 + i] = dd4[i];

  float widths[5], heights[5], cw[6], ch[6];
  // widths = softmax(w2) -> affine -> cumsum -> knots
  {
    float mx = w2[0];
    #pragma unroll
    for (int i = 1; i < 5; ++i) mx = fmaxf(mx, w2[i]);
    float s = 0.f;
    #pragma unroll
    for (int i = 0; i < 5; ++i) { widths[i] = __expf(w2[i] - mx); s += widths[i]; }
    const float invs = __builtin_amdgcn_rcpf(s);
    #pragma unroll
    for (int i = 0; i < 5; ++i)
      widths[i] = 0.001f + (1.0f - 0.001f * 5.0f) * (widths[i] * invs);
    float c = 0.f;
    cw[0] = -1.0f;
    #pragma unroll
    for (int i = 0; i < 5; ++i) { c += widths[i]; cw[i + 1] = 2.0f * c - 1.0f; }
    cw[5] = 1.0f;
    #pragma unroll
    for (int i = 0; i < 5; ++i) widths[i] = cw[i + 1] - cw[i];
  }
  {
    float mx = h2[0];
    #pragma unroll
    for (int i = 1; i < 5; ++i) mx = fmaxf(mx, h2[i]);
    float s = 0.f;
    #pragma unroll
    for (int i = 0; i < 5; ++i) { heights[i] = __expf(h2[i] - mx); s += heights[i]; }
    const float invs = __builtin_amdgcn_rcpf(s);
    #pragma unroll
    for (int i = 0; i < 5; ++i)
      heights[i] = 0.001f + (1.0f - 0.001f * 5.0f) * (heights[i] * invs);
    float c = 0.f;
    ch[0] = -1.0f;
    #pragma unroll
    for (int i = 0; i < 5; ++i) { c += heights[i]; ch[i + 1] = 2.0f * c - 1.0f; }
    ch[5] = 1.0f;
    #pragma unroll
    for (int i = 0; i < 5; ++i) heights[i] = ch[i + 1] - ch[i];
  }

  float dv[6];
  #pragma unroll
  for (int i = 0; i < 6; ++i) dv[i] = 0.001f + fast_softplusf(dpad[i]);

  const float xc = fminf(fmaxf(xv, -1.0f), 1.0f);
  int cnt = 0;
  #pragma unroll
  for (int i = 0; i < 6; ++i) cnt += (xc >= cw[i]) ? 1 : 0;
  int idx = cnt - 1;
  idx = (idx < 0) ? 0 : ((idx > 4) ? 4 : idx);

  const float in_cw = cw[idx];
  const float in_w  = widths[idx];
  const float in_ch = ch[idx];
  const float in_h  = heights[idx];
  const float delta = in_h / in_w;
  const float d_k   = dv[idx];
  const float d_k1  = dv[idx + 1];

  const float theta = (xc - in_cw) / in_w;
  const float tom   = theta * (1.0f - theta);
  const float num   = in_h * (delta * theta * theta + d_k * tom);
  const float den   = delta + (d_k + d_k1 - 2.0f * delta) * tom;
  const float out_in = in_ch + num / den;
  const float omt   = 1.0f - theta;
  const float dnum  = delta * delta *
      (d_k1 * theta * theta + 2.0f * delta * tom + d_k * omt * omt);
  const float ld_in = __logf(dnum) - 2.0f * __logf(den);

  const bool inside = (xv >= -1.0f) && (xv <= 1.0f);
  out[(size_t)m * IN_DIM + d]                          = inside ? out_in : xv;
  out[(size_t)BATCH * IN_DIM + (size_t)m * IN_DIM + d] = inside ? ld_in : 0.0f;
}

// ---------------------------------------------------------------------------
// Weight packing kernels (f32 -> padded f16 layout in workspace)
// ---------------------------------------------------------------------------
__global__ void fill_zero_u32(unsigned int* p, int n) {
  int i = blockIdx.x * 256 + threadIdx.x;
  if (i < n) p[i] = 0u;
}

__global__ void cast_copy_pad(const float* __restrict__ src, _Float16* __restrict__ dst,
                              int rows, int cols, int dst_stride, int dst_off) {
  int i = blockIdx.x * 256 + threadIdx.x;
  if (i >= rows * cols) return;
  int r = i / cols, c = i % cols;
  dst[(size_t)r * dst_stride + dst_off + c] = (_Float16)src[i];
}

__global__ void copy_f32(const float* __restrict__ src, float* __restrict__ dst, int n) {
  int i = blockIdx.x * 256 + threadIdx.x;
  if (i < n) dst[i] = src[i];
}

// ---------------------------------------------------------------------------
// Launcher
// ---------------------------------------------------------------------------
extern "C" void kernel_launch(void* const* d_in, const int* in_sizes, int n_in,
                              void* d_out, int out_size, void* d_ws, size_t ws_size,
                              hipStream_t stream) {
  (void)in_sizes; (void)n_in; (void)out_size; (void)ws_size;

  const float* x     = (const float*)d_in[0];
  const float* cond  = (const float*)d_in[1];
  const float* initp = (const float*)d_in[2];
  float*       out   = (float*)d_out;

  // Workspace layout
  const size_t PBUF_FLOATS = 7ull * BATCH * 16;           // p outputs, padded to 16
  float*     p_buf = (float*)d_ws;
  _Float16*  wpack = (_Float16*)((char*)d_ws + PBUF_FLOATS * sizeof(float));
  float*     bpack = (float*)((char*)wpack + 7ull * WPD * sizeof(_Float16));

  // Zero the packed-weight + bias region (covers all padding)
  {
    const size_t bytes = 7ull * WPD * sizeof(_Float16) + 7ull * BPD * sizeof(float);
    const int nwords = (int)(bytes / 4);
    fill_zero_u32<<<(nwords + 255) / 256, 256, 0, stream>>>((unsigned int*)wpack, nwords);
  }

  // Pack weights for each dim (params flattened: per dim, per layer: w, b, cw)
  for (int dd = 0; dd < 7; ++dd) {
    const int dim = dd + 1;
    const int base = 3 + dd * 12;
    const float* w0  = (const float*)d_in[base + 0];
    const float* b0  = (const float*)d_in[base + 1];
    const float* cw0 = (const float*)d_in[base + 2];
    const float* w1  = (const float*)d_in[base + 3];
    const float* b1  = (const float*)d_in[base + 4];
    const float* cw1 = (const float*)d_in[base + 5];
    const float* w2  = (const float*)d_in[base + 6];
    const float* b2  = (const float*)d_in[base + 7];
    const float* cw2 = (const float*)d_in[base + 8];
    const float* w3  = (const float*)d_in[base + 9];
    const float* b3  = (const float*)d_in[base + 10];
    const float* cw3 = (const float*)d_in[base + 11];

    _Float16* wb = wpack + (size_t)dd * WPD;
    float*    bp = bpack + (size_t)dd * BPD;

    auto blocks = [](int n) { return (n + 255) / 256; };

    // layer0 fused into K=32 tile: [w0 | cw0 | 0]
    cast_copy_pad<<<blocks(256 * dim), 256, 0, stream>>>(w0,  wb + OFF_B0, 256, dim, 32, 0);
    cast_copy_pad<<<blocks(256 * 6),   256, 0, stream>>>(cw0, wb + OFF_B0, 256, 6,  32, dim);
    copy_f32<<<1, 256, 0, stream>>>(b0, bp, 256);

    cast_copy_pad<<<blocks(256 * 256), 256, 0, stream>>>(w1,  wb + OFF_W1,  256, 256, 256, 0);
    cast_copy_pad<<<blocks(256 * 6),   256, 0, stream>>>(cw1, wb + OFF_CW1, 256, 6,   32,  0);
    copy_f32<<<1, 256, 0, stream>>>(b1, bp + 256, 256);

    cast_copy_pad<<<blocks(256 * 256), 256, 0, stream>>>(w2,  wb + OFF_W2,  256, 256, 256, 0);
    cast_copy_pad<<<blocks(256 * 6),   256, 0, stream>>>(cw2, wb + OFF_CW2, 256, 6,   32,  0);
    copy_f32<<<1, 256, 0, stream>>>(b2, bp + 512, 256);

    cast_copy_pad<<<blocks(14 * 256),  256, 0, stream>>>(w3,  wb + OFF_W3,  14, 256, 256, 0);
    cast_copy_pad<<<blocks(14 * 6),    256, 0, stream>>>(cw3, wb + OFF_CW3, 14, 6,   32,  0);
    copy_f32<<<1, 64, 0, stream>>>(b3, bp + 768, 14);
  }

  // Fused MLPs: 1024 row-tiles x 7 dims
  {
    dim3 grid(BATCH / MTILE, 7, 1);
    nsf_mlp_kernel<<<grid, 256, 0, stream>>>(x, cond, wpack, bpack, p_buf);
  }

  // Spline transform for all 8 dims
  {
    const int total = BATCH * IN_DIM;
    nsf_spline_kernel<<<total / 256, 256, 0, stream>>>(x, initp, p_buf, out);
  }
}
